// PGSSTB_12850542150449
// MI455X (gfx1250) — matrix operand order repaired and checked by reading, not compiled, not run on hardware
//
#include <hip/hip_runtime.h>
#include <cstdint>

// ---------------- constants ----------------
#define B_      4
#define C_      192
#define HW_     128
#define N_      16384          // HW_*HW_
#define HEADS_  6
#define HD_     32
#define T3C_    576            // 3*C_
#define HID_    768            // 4*C_ (MLP hidden after gate)
#define FC1O_   1536           // 2*HID_
#define CR_     24
#define PLEN_   128
#define MTOT_   65536          // B_*N_

typedef __bf16 bf16;
typedef __attribute__((ext_vector_type(16))) __bf16 v16bf;
typedef __attribute__((ext_vector_type(8)))  float  v8f;

union TileU  { v16bf v; uint32_t u[8]; };
union BfPair { uint32_t u; __bf16 h[2]; };

__device__ inline v8f vzero() {
  v8f z;
#pragma unroll
  for (int i = 0; i < 8; ++i) z[i] = 0.0f;
  return z;
}

__device__ inline uint32_t packbf(float lo, float hi) {
  BfPair p; p.h[0] = (__bf16)lo; p.h[1] = (__bf16)hi; return p.u;
}

// A-operand tile (16 rows x 32 K), ISA 7.12.2 "16-bit A-Matrix 16x32" layout.
// lanes 0-15: V0..3 = K{0..7} pairs, V4..7 = K{16..23}; lanes 16-31: K{8..15},{24..31}
__device__ inline v16bf load_tileA(const bf16* base, int ld, int lane) {
  const int r = lane & 15, half = lane >> 4;
  const bf16* p = base + (size_t)r * ld;
  TileU t;
#pragma unroll
  for (int j = 0; j < 4; ++j) t.u[j]     = *(const uint32_t*)(p + half * 8 + 2 * j);
#pragma unroll
  for (int j = 0; j < 4; ++j) t.u[4 + j] = *(const uint32_t*)(p + 16 + half * 8 + 2 * j);
  return t.v;
}

// B-operand tile (32 K x 16 cols) loaded column-major: source is Bop^T row-major
// (i.e. weight rows). lanes 0-15 hold K{0..7},{16..23}, lanes 16-31 hold K{8..15},{24..31} (same as A).
__device__ inline v16bf load_tileB(const bf16* base, int ld, int lane) {
  const int n = lane & 15, kh = lane >> 4;
  const bf16* p = base + (size_t)n * ld + kh * 8;
  TileU t;
#pragma unroll
  for (int j = 0; j < 8; ++j) t.u[j] = *(const uint32_t*)(p + (j >> 2) * 16 + 2 * (j & 3));
  return t.v;
}

__device__ inline v8f wmma_bf(v16bf a, v16bf b, v8f c) {
  return __builtin_amdgcn_wmma_f32_16x16x32_bf16(false, a, false, b, (short)0, c, false, false);
}

// ---------------- f32 -> bf16 weight conversion ----------------
__global__ void k_cvt(const float* __restrict__ in, bf16* __restrict__ out, int n) {
  int i = blockIdx.x * 256 + threadIdx.x;
  if (i < n) out[i] = (bf16)in[i];
}

// ---------------- LN1 + token-major copy + windowed layout ----------------
// wave (32 lanes) per token; lane handles 6 channels
__global__ void __launch_bounds__(256) k_prep(const float* __restrict__ x4,
                                              const float* __restrict__ g,
                                              const float* __restrict__ bb,
                                              bf16* __restrict__ tok,
                                              bf16* __restrict__ xw) {
  const int lane = threadIdx.x & 31;
  const int t = blockIdx.x * 8 + (threadIdx.x >> 5);
  const int b = t >> 14, n = t & (N_ - 1);
  const float* src = x4 + (size_t)b * C_ * N_ + n;
  float v[6];
  float s = 0.f;
#pragma unroll
  for (int j = 0; j < 6; ++j) { v[j] = src[(size_t)(lane + 32 * j) * N_]; s += v[j]; }
#pragma unroll
  for (int off = 1; off < 32; off <<= 1) s += __shfl_xor(s, off, 32);
  const float mean = s * (1.f / C_);
  float var = 0.f;
#pragma unroll
  for (int j = 0; j < 6; ++j) { float d = v[j] - mean; var += d * d; }
#pragma unroll
  for (int off = 1; off < 32; off <<= 1) var += __shfl_xor(var, off, 32);
  const float rstd = rsqrtf(var * (1.f / C_) + 1e-5f);
  const int hh = n >> 7, ww = n & 127;
  const int win = (b << 8) | ((hh >> 3) << 4) | (ww >> 3);
  const int wrow = (win << 6) | ((hh & 7) << 3) | (ww & 7);
#pragma unroll
  for (int j = 0; j < 6; ++j) {
    int c = lane + 32 * j;
    tok[(size_t)t * C_ + c] = (bf16)v[j];
    xw[(size_t)wrow * C_ + c] = (bf16)((v[j] - mean) * rstd * g[c] + bb[c]);
  }
}

// ---------------- generic WMMA GEMM: out = A[MxK] * W[NoutxK]^T (+bias) ----------------
// block 256 = 8 waves; block tile 128 rows x 64 cols; wave tile 32x32
enum GemmMode { OUT_F32 = 0, OUT_BF16 = 1, OUT_XSP = 2 };

template <int MODE, bool HAS_BIAS>
__global__ void __launch_bounds__(256) k_gemm(const bf16* __restrict__ A,
                                              const bf16* __restrict__ W,
                                              const float* __restrict__ bias,
                                              void* __restrict__ out,
                                              int M, int Nout, int K) {
  const int lane = threadIdx.x & 31;
  const int wid = threadIdx.x >> 5;
  const int row0 = blockIdx.x * 128 + (wid & 3) * 32;
  const int col0 = blockIdx.y * 64 + (wid >> 2) * 32;
  if (row0 >= M || col0 >= Nout) return;
  v8f a00 = vzero(), a01 = vzero(), a10 = vzero(), a11 = vzero();
  for (int k = 0; k < K; k += 32) {
    v16bf ta0 = load_tileA(A + (size_t)row0 * K + k, K, lane);
    v16bf ta1 = load_tileA(A + (size_t)(row0 + 16) * K + k, K, lane);
    v16bf tb0 = load_tileB(W + (size_t)col0 * K + k, K, lane);
    v16bf tb1 = load_tileB(W + (size_t)(col0 + 16) * K + k, K, lane);
    a00 = wmma_bf(ta0, tb0, a00);
    a01 = wmma_bf(ta0, tb1, a01);
    a10 = wmma_bf(ta1, tb0, a10);
    a11 = wmma_bf(ta1, tb1, a11);
  }
  const int r = lane & 15, half = lane >> 4;
#pragma unroll
  for (int at = 0; at < 2; ++at)
#pragma unroll
    for (int bt = 0; bt < 2; ++bt) {
      v8f acc = (at == 0) ? (bt == 0 ? a00 : a01) : (bt == 0 ? a10 : a11);
#pragma unroll
      for (int i = 0; i < 8; ++i) {
        int row = row0 + at * 16 + i + 8 * half;
        int col = col0 + bt * 16 + r;
        float v = acc[i];
        if (HAS_BIAS) v += bias[col];
        if (MODE == OUT_F32) {
          ((float*)out)[(size_t)row * Nout + col] = v;
        } else if (MODE == OUT_BF16) {
          ((bf16*)out)[(size_t)row * Nout + col] = (bf16)v;
        } else {  // OUT_XSP: windowed row -> token order (for swin proj)
          int win = row >> 6, t = row & 63;
          int b = win >> 8, wh = (win >> 4) & 15, ww = win & 15;
          int hh = wh * 8 + (t >> 3), wc = ww * 8 + (t & 7);
          int n = hh * HW_ + wc;
          ((float*)out)[((size_t)b * N_ + n) * C_ + col] = v;
        }
      }
    }
}

// ---------------- swin windowed attention, one (window, head) per block ----------------
// block 128 = 4 waves; wave handles a 16-row band of the 64x64 score matrix
__global__ void __launch_bounds__(128) k_swin(const bf16* __restrict__ qkv,
                                              const float* __restrict__ rpb,
                                              bf16* __restrict__ attno) {
  const int win = blockIdx.x / HEADS_, h = blockIdx.x % HEADS_;
  const int wid = threadIdx.x >> 5, lane = threadIdx.x & 31;
  __shared__ __align__(16) bf16 Vt[32][64];      // V transposed: Vt[d][j]
  __shared__ float Ps[4][16][64];                // per-wave softmaxed P band
  for (int idx = threadIdx.x; idx < 2048; idx += 128) {
    int j = idx >> 5, d = idx & 31;
    Vt[d][j] = qkv[(size_t)(win * 64 + j) * T3C_ + 384 + h * HD_ + d];
  }
  __syncthreads();
  const int m0 = wid * 16;
  const int r = lane & 15, half = lane >> 4;
  // S = Q * K^T  (K-dim = 32, single wmma step per col tile)
  v16bf aq = load_tileA(qkv + (size_t)(win * 64 + m0) * T3C_ + h * HD_, T3C_, lane);
  v8f s[4];
#pragma unroll
  for (int jt = 0; jt < 4; ++jt) {
    v16bf bk = load_tileB(qkv + (size_t)(win * 64 + jt * 16) * T3C_ + 192 + h * HD_, T3C_, lane);
    s[jt] = wmma_bf(aq, bk, vzero());
  }
  // scale + relative position bias + row softmax (rows live in 16-lane halves)
#pragma unroll
  for (int i = 0; i < 8; ++i) {
    const int m = m0 + i + 8 * half;
    const int ty = (m >> 3) & 7, tx = m & 7;
    float mx = -1e30f;
#pragma unroll
    for (int jt = 0; jt < 4; ++jt) {
      int j = jt * 16 + r;
      int bidx = (ty - (j >> 3) + 7) * 15 + (tx - (j & 7) + 7);
      float v = s[jt][i] * 0.17677669529663687f + rpb[bidx * HEADS_ + h];
      s[jt][i] = v; mx = fmaxf(mx, v);
    }
    for (int off = 1; off < 16; off <<= 1) mx = fmaxf(mx, __shfl_xor(mx, off, 32));
    float sum = 0.f;
#pragma unroll
    for (int jt = 0; jt < 4; ++jt) { float e = __expf(s[jt][i] - mx); s[jt][i] = e; sum += e; }
    for (int off = 1; off < 16; off <<= 1) sum += __shfl_xor(sum, off, 32);
    float inv = 1.f / sum;
#pragma unroll
    for (int jt = 0; jt < 4; ++jt) s[jt][i] *= inv;
  }
  // stage P to LDS (re-layout D->A operand)
#pragma unroll
  for (int i = 0; i < 8; ++i)
#pragma unroll
    for (int jt = 0; jt < 4; ++jt)
      Ps[wid][i + 8 * half][jt * 16 + r] = s[jt][i];
  __syncthreads();
  // O = P * V  (K = 64 -> two 32-wide steps; 2 col tiles of 16)
  v8f o0 = vzero(), o1 = vzero();
#pragma unroll
  for (int ks = 0; ks < 2; ++ks) {
    TileU ta;
    const float* prow = &Ps[wid][r][ks * 32];
#pragma unroll
    for (int j = 0; j < 4; ++j)
      ta.u[j] = packbf(prow[half * 8 + 2 * j], prow[half * 8 + 2 * j + 1]);
#pragma unroll
    for (int j = 0; j < 4; ++j)
      ta.u[4 + j] = packbf(prow[16 + half * 8 + 2 * j], prow[16 + half * 8 + 2 * j + 1]);
    v16bf vb0 = load_tileB(&Vt[0][0] + ks * 32, 64, lane);
    v16bf vb1 = load_tileB(&Vt[16][0] + ks * 32, 64, lane);
    o0 = wmma_bf(ta.v, vb0, o0);
    o1 = wmma_bf(ta.v, vb1, o1);
  }
#pragma unroll
  for (int i = 0; i < 8; ++i) {
    int row = win * 64 + m0 + i + 8 * half;
    attno[(size_t)row * C_ + h * HD_ + r] = (bf16)o0[i];
    attno[(size_t)row * C_ + h * HD_ + 16 + r] = (bf16)o1[i];
  }
}

// ---------------- depthwise 3x3 conv on token-major qkv2 ----------------
// q,k -> channel-major (for Gram WMMA operands); v -> token-major
__global__ void __launch_bounds__(576) k_dwconv(const bf16* __restrict__ x,
                                                const float* __restrict__ w,
                                                bf16* __restrict__ qk,
                                                bf16* __restrict__ v2t) {
  const int t = blockIdx.x;
  const int c = threadIdx.x;
  const int b = t >> 14, n = t & (N_ - 1);
  const int hh = n >> 7, ww = n & 127;
  float acc = 0.f;
#pragma unroll
  for (int dy = -1; dy <= 1; ++dy)
#pragma unroll
    for (int dx = -1; dx <= 1; ++dx) {
      int y = hh + dy, xx = ww + dx;
      if (y >= 0 && y < HW_ && xx >= 0 && xx < HW_) {
        int nn = y * HW_ + xx;
        acc += (float)x[(size_t)(b * N_ + nn) * T3C_ + c] * w[c * 9 + (dy + 1) * 3 + (dx + 1)];
      }
    }
  if (c < 384) qk[((size_t)b * 384 + c) * N_ + n] = (bf16)acc;
  else         v2t[((size_t)(b * N_ + n)) * C_ + (c - 384)] = (bf16)acc;
}

// ---------------- row L2 norms of q2/k2 over N ----------------
__global__ void __launch_bounds__(256) k_rownorm(const bf16* __restrict__ qk,
                                                 float* __restrict__ norms) {
  __shared__ float sm[256];
  const int row = blockIdx.x;  // b*384 + ch
  const bf16* p = qk + (size_t)row * N_;
  float s = 0.f;
  for (int n = threadIdx.x; n < N_; n += 256) { float v = (float)p[n]; s += v * v; }
  sm[threadIdx.x] = s; __syncthreads();
  for (int st = 128; st > 0; st >>= 1) {
    if (threadIdx.x < st) sm[threadIdx.x] += sm[threadIdx.x + st];
    __syncthreads();
  }
  if (threadIdx.x == 0) norms[row] = fmaxf(sqrtf(sm[0]), 1e-12f);
}

// ---------------- Gram matrix a2 = softmax((q^ k^T)*temp), per (b,head) ----------------
// block 256 = 8 waves splitting K=16384; LDS atomic reduce; softmax by 32 lanes
__global__ void __launch_bounds__(256) k_a2(const bf16* __restrict__ qk,
                                            const float* __restrict__ norms,
                                            const float* __restrict__ temp,
                                            bf16* __restrict__ a2bf) {
  const int b = blockIdx.x / HEADS_, h = blockIdx.x % HEADS_;
  const int wid = threadIdx.x >> 5, lane = threadIdx.x & 31;
  __shared__ float sA[32][33];
  for (int i = threadIdx.x; i < 32 * 33; i += 256) (&sA[0][0])[i] = 0.f;
  __syncthreads();
  const bf16* qb = qk + ((size_t)b * 384 + h * HD_) * N_;
  const bf16* kb = qk + ((size_t)b * 384 + 192 + h * HD_) * N_;
  v8f a00 = vzero(), a01 = vzero(), a10 = vzero(), a11 = vzero();
  for (int k0 = wid * 2048; k0 < (wid + 1) * 2048; k0 += 32) {
    v16bf ta0 = load_tileA(qb + k0, N_, lane);
    v16bf ta1 = load_tileA(qb + (size_t)16 * N_ + k0, N_, lane);
    v16bf tb0 = load_tileB(kb + k0, N_, lane);
    v16bf tb1 = load_tileB(kb + (size_t)16 * N_ + k0, N_, lane);
    a00 = wmma_bf(ta0, tb0, a00);
    a01 = wmma_bf(ta0, tb1, a01);
    a10 = wmma_bf(ta1, tb0, a10);
    a11 = wmma_bf(ta1, tb1, a11);
  }
  const int r = lane & 15, half = lane >> 4;
#pragma unroll
  for (int at = 0; at < 2; ++at)
#pragma unroll
    for (int bt = 0; bt < 2; ++bt) {
      v8f acc = (at == 0) ? (bt == 0 ? a00 : a01) : (bt == 0 ? a10 : a11);
#pragma unroll
      for (int i = 0; i < 8; ++i)
        atomicAdd(&sA[at * 16 + i + 8 * half][bt * 16 + r], acc[i]);
    }
  __syncthreads();
  if (threadIdx.x < 32) {
    const int c = threadIdx.x;
    const float nq = norms[b * 384 + h * HD_ + c];
    const float tt = temp[h];
    float row[32]; float mx = -1e30f;
#pragma unroll
    for (int d = 0; d < 32; ++d) {
      float v = sA[c][d] * tt / (nq * norms[b * 384 + 192 + h * HD_ + d]);
      row[d] = v; mx = fmaxf(mx, v);
    }
    float sum = 0.f;
#pragma unroll
    for (int d = 0; d < 32; ++d) { row[d] = __expf(row[d] - mx); sum += row[d]; }
    float inv = 1.f / sum;
#pragma unroll
    for (int d = 0; d < 32; ++d)
      a2bf[(size_t)(b * HEADS_ + h) * 1024 + c * 32 + d] = (bf16)(row[d] * inv);
  }
}

// ---------------- o2^T[n][c] = sum_d v2t[n][d] * a2[c][d]  (per b,head) ----------------
__global__ void __launch_bounds__(128) k_o2t(const bf16* __restrict__ v2t,
                                             const bf16* __restrict__ a2bf,
                                             bf16* __restrict__ ot) {
  const int bh = blockIdx.x >> 7;            // 128 row-blocks per (b,h)
  const int rb = blockIdx.x & 127;
  const int b = bh / HEADS_, h = bh % HEADS_;
  const int wid = threadIdx.x >> 5, lane = threadIdx.x & 31;
  const int n0 = rb * 128 + wid * 32;
  const bf16* Ab = v2t + ((size_t)b * N_ + n0) * C_ + h * HD_;
  const bf16* Bb = a2bf + (size_t)(b * HEADS_ + h) * 1024;
  v16bf ta0 = load_tileA(Ab, C_, lane);
  v16bf ta1 = load_tileA(Ab + 16 * C_, C_, lane);
  v16bf tb0 = load_tileB(Bb, 32, lane);
  v16bf tb1 = load_tileB(Bb + 16 * 32, 32, lane);
  v8f a00 = wmma_bf(ta0, tb0, vzero());
  v8f a01 = wmma_bf(ta0, tb1, vzero());
  v8f a10 = wmma_bf(ta1, tb0, vzero());
  v8f a11 = wmma_bf(ta1, tb1, vzero());
  const int r = lane & 15, half = lane >> 4;
#pragma unroll
  for (int at = 0; at < 2; ++at)
#pragma unroll
    for (int bt = 0; bt < 2; ++bt) {
      v8f acc = (at == 0) ? (bt == 0 ? a00 : a01) : (bt == 0 ? a10 : a11);
#pragma unroll
      for (int i = 0; i < 8; ++i) {
        int n = n0 + at * 16 + i + 8 * half;
        ot[((size_t)b * N_ + n) * C_ + h * HD_ + bt * 16 + r] = (bf16)acc[i];
      }
    }
}

// ---------------- mean over N of spec ----------------
__global__ void __launch_bounds__(256) k_mean(const float* __restrict__ spec,
                                              float* __restrict__ xm) {
  __shared__ float sm[256];
  const int b = blockIdx.x / C_, c = blockIdx.x % C_;
  float s = 0.f;
  for (int n = threadIdx.x; n < N_; n += 256) s += spec[((size_t)b * N_ + n) * C_ + c];
  sm[threadIdx.x] = s; __syncthreads();
  for (int st = 128; st > 0; st >>= 1) {
    if (threadIdx.x < st) sm[threadIdx.x] += sm[threadIdx.x + st];
    __syncthreads();
  }
  if (threadIdx.x == 0) xm[b * C_ + c] = sm[0] * (1.f / N_);
}

// ---------------- tiny prompt-guided branch -> per-(b,c) gate vector ----------------
__global__ void __launch_bounds__(192) k_pg(const float* __restrict__ xm,
                                            const float* __restrict__ promptw,
                                            const float* __restrict__ downw,
                                            const float* __restrict__ promptp,
                                            const float* __restrict__ qw,
                                            const float* __restrict__ kvw,
                                            const float* __restrict__ projw,
                                            const float* __restrict__ projb,
                                            const float* __restrict__ upw,
                                            float* __restrict__ pgvec) {
  __shared__ float s_xm[C_], s_pw[PLEN_], s_xd[CR_], s_sp[CR_];
  __shared__ float s_qp[CR_], s_kp[CR_], s_vp[CR_], s_op[CR_], s_op2[CR_];
  const int tid = threadIdx.x;
  for (int b = 0; b < B_; ++b) {
    s_xm[tid] = xm[b * C_ + tid];
    __syncthreads();
    if (tid < PLEN_) {
      float a = 0.f;
      for (int c = 0; c < C_; ++c) a += s_xm[c] * promptw[tid * C_ + c];
      s_pw[tid] = a;
    }
    __syncthreads();
    if (tid == 0) {
      float mx = -1e30f;
      for (int p = 0; p < PLEN_; ++p) mx = fmaxf(mx, s_pw[p]);
      float sm = 0.f;
      for (int p = 0; p < PLEN_; ++p) { s_pw[p] = __expf(s_pw[p] - mx); sm += s_pw[p]; }
      float inv = 1.f / sm;
      for (int p = 0; p < PLEN_; ++p) s_pw[p] *= inv;
    }
    __syncthreads();
    if (tid < CR_) {
      float a = 0.f, sp = 0.f;
      for (int c = 0; c < C_; ++c) a += s_xm[c] * downw[tid * C_ + c];
      for (int p = 0; p < PLEN_; ++p) sp += s_pw[p] * promptp[p * CR_ + tid];
      s_xd[tid] = a; s_sp[tid] = sp;
    }
    __syncthreads();
    if (tid < CR_) {
      float q = 0.f, k = 0.f, v = 0.f;
      for (int cr = 0; cr < CR_; ++cr) {
        q += s_sp[cr] * qw[tid * CR_ + cr];
        k += s_xd[cr] * kvw[tid * CR_ + cr];
        v += s_xd[cr] * kvw[(CR_ + tid) * CR_ + cr];
      }
      s_qp[tid] = q; s_kp[tid] = k; s_vp[tid] = v;
    }
    __syncthreads();
    if (tid < CR_) {
      const float scale = 0.20412414523193154f;  // 24^-0.5
      float row[CR_]; float mx = -1e30f;
      for (int d = 0; d < CR_; ++d) { row[d] = s_qp[tid] * s_kp[d] * scale; mx = fmaxf(mx, row[d]); }
      float sm = 0.f;
      for (int d = 0; d < CR_; ++d) { row[d] = __expf(row[d] - mx); sm += row[d]; }
      float inv = 1.f / sm, o = 0.f;
      for (int d = 0; d < CR_; ++d) o += row[d] * inv * s_vp[d];
      s_op[tid] = o;
    }
    __syncthreads();
    if (tid < CR_) {
      float a = projb[tid];
      for (int d = 0; d < CR_; ++d) a += s_op[d] * projw[tid * CR_ + d];
      s_op2[tid] = a;
    }
    __syncthreads();
    {
      float a = 0.f;
      for (int cr = 0; cr < CR_; ++cr) a += s_op2[cr] * upw[tid * CR_ + cr];
      pgvec[b * C_ + tid] = a;
    }
    __syncthreads();
  }
}

// ---------------- residual merge + LN2 (x written in-place over spec) ----------------
__global__ void __launch_bounds__(256) k_merge(const float* __restrict__ x4,
                                               const float* __restrict__ xsp,
                                               float* __restrict__ specx,
                                               const float* __restrict__ pgvec,
                                               const float* __restrict__ g,
                                               const float* __restrict__ bb,
                                               bf16* __restrict__ xln2) {
  const int lane = threadIdx.x & 31;
  const int t = blockIdx.x * 8 + (threadIdx.x >> 5);
  const int b = t >> 14, n = t & (N_ - 1);
  float v[6]; float s = 0.f;
#pragma unroll
  for (int j = 0; j < 6; ++j) {
    int c = lane + 32 * j;
    float sc = x4[((size_t)b * C_ + c) * N_ + n];
    float xv = sc + xsp[(size_t)t * C_ + c] + pgvec[b * C_ + c] * specx[(size_t)t * C_ + c];
    v[j] = xv; s += xv;
  }
#pragma unroll
  for (int off = 1; off < 32; off <<= 1) s += __shfl_xor(s, off, 32);
  const float mean = s * (1.f / C_);
  float var = 0.f;
#pragma unroll
  for (int j = 0; j < 6; ++j) { float d = v[j] - mean; var += d * d; }
#pragma unroll
  for (int off = 1; off < 32; off <<= 1) var += __shfl_xor(var, off, 32);
  const float rstd = rsqrtf(var * (1.f / C_) + 1e-5f);
#pragma unroll
  for (int j = 0; j < 6; ++j) {
    int c = lane + 32 * j;
    specx[(size_t)t * C_ + c] = v[j];
    xln2[(size_t)t * C_ + c] = (bf16)((v[j] - mean) * rstd * g[c] + bb[c]);
  }
}

// ---------------- fc1 with fused gate: gated = (h_a+b) * gelu(h_g+b) ----------------
__global__ void __launch_bounds__(256) k_fc1gate(const bf16* __restrict__ A,
                                                 const bf16* __restrict__ W,
                                                 const float* __restrict__ bias,
                                                 bf16* __restrict__ gated) {
  const int lane = threadIdx.x & 31;
  const int wid = threadIdx.x >> 5;
  const int row0 = blockIdx.x * 128 + (wid & 3) * 32;
  const int col0 = blockIdx.y * 64 + (wid >> 2) * 32;
  v8f aA[2][2], aG[2][2];
#pragma unroll
  for (int i = 0; i < 2; ++i)
#pragma unroll
    for (int j = 0; j < 2; ++j) { aA[i][j] = vzero(); aG[i][j] = vzero(); }
  for (int k = 0; k < C_; k += 32) {
    v16bf ta0 = load_tileA(A + (size_t)row0 * C_ + k, C_, lane);
    v16bf ta1 = load_tileA(A + (size_t)(row0 + 16) * C_ + k, C_, lane);
#pragma unroll
    for (int bt = 0; bt < 2; ++bt) {
      v16bf bA = load_tileB(W + (size_t)(col0 + bt * 16) * C_ + k, C_, lane);
      v16bf bG = load_tileB(W + (size_t)(col0 + HID_ + bt * 16) * C_ + k, C_, lane);
      aA[0][bt] = wmma_bf(ta0, bA, aA[0][bt]);
      aA[1][bt] = wmma_bf(ta1, bA, aA[1][bt]);
      aG[0][bt] = wmma_bf(ta0, bG, aG[0][bt]);
      aG[1][bt] = wmma_bf(ta1, bG, aG[1][bt]);
    }
  }
  const int r = lane & 15, half = lane >> 4;
#pragma unroll
  for (int at = 0; at < 2; ++at)
#pragma unroll
    for (int bt = 0; bt < 2; ++bt)
#pragma unroll
      for (int i = 0; i < 8; ++i) {
        int row = row0 + at * 16 + i + 8 * half;
        int col = col0 + bt * 16 + r;
        float av = aA[at][bt][i] + bias[col];
        float gv = aG[at][bt][i] + bias[col + HID_];
        float ge = 0.5f * gv * (1.f + erff(gv * 0.7071067811865475f));
        gated[(size_t)row * HID_ + col] = (bf16)(av * ge);
      }
}

// ---------------- fc2 + bias + residual + transpose to (B,C,H,W) ----------------
__global__ void __launch_bounds__(256) k_fc2(const bf16* __restrict__ A,
                                             const bf16* __restrict__ W,
                                             const float* __restrict__ bias,
                                             const float* __restrict__ xres,
                                             float* __restrict__ out) {
  const int lane = threadIdx.x & 31;
  const int wid = threadIdx.x >> 5;
  const int row0 = blockIdx.x * 128 + (wid & 3) * 32;
  const int col0 = blockIdx.y * 64 + (wid >> 2) * 32;
  v8f a00 = vzero(), a01 = vzero(), a10 = vzero(), a11 = vzero();
  for (int k = 0; k < HID_; k += 32) {
    v16bf ta0 = load_tileA(A + (size_t)row0 * HID_ + k, HID_, lane);
    v16bf ta1 = load_tileA(A + (size_t)(row0 + 16) * HID_ + k, HID_, lane);
    v16bf tb0 = load_tileB(W + (size_t)col0 * HID_ + k, HID_, lane);
    v16bf tb1 = load_tileB(W + (size_t)(col0 + 16) * HID_ + k, HID_, lane);
    a00 = wmma_bf(ta0, tb0, a00);
    a01 = wmma_bf(ta0, tb1, a01);
    a10 = wmma_bf(ta1, tb0, a10);
    a11 = wmma_bf(ta1, tb1, a11);
  }
  const int r = lane & 15, half = lane >> 4;
#pragma unroll
  for (int at = 0; at < 2; ++at)
#pragma unroll
    for (int bt = 0; bt < 2; ++bt) {
      v8f acc = (at == 0) ? (bt == 0 ? a00 : a01) : (bt == 0 ? a10 : a11);
#pragma unroll
      for (int i = 0; i < 8; ++i) {
        int row = row0 + at * 16 + i + 8 * half;
        int col = col0 + bt * 16 + r;
        int b = row >> 14, n = row & (N_ - 1);
        out[((size_t)b * C_ + col) * N_ + n] =
            xres[(size_t)row * C_ + col] + acc[i] + bias[col];
      }
    }
}

// ---------------- launcher ----------------
extern "C" void kernel_launch(void* const* d_in, const int* in_sizes, int n_in,
                              void* d_out, int out_size, void* d_ws, size_t ws_size,
                              hipStream_t stream) {
  (void)in_sizes; (void)n_in; (void)out_size; (void)ws_size;
  const float* in_put    = (const float*)d_in[0];
  const float* ln1_g     = (const float*)d_in[1];
  const float* ln1_b     = (const float*)d_in[2];
  const float* ln2_g     = (const float*)d_in[3];
  const float* ln2_b     = (const float*)d_in[4];
  const float* qkv_w     = (const float*)d_in[5];
  const float* qkv_b     = (const float*)d_in[6];
  const float* proj_w    = (const float*)d_in[7];
  const float* proj_b    = (const float*)d_in[8];
  const float* rpb_table = (const float*)d_in[9];
  const float* gs_qkv_w  = (const float*)d_in[10];
  const float* gs_dw_w   = (const float*)d_in[11];
  const float* gs_proj_w = (const float*)d_in[12];
  const float* temperature = (const float*)d_in[13];
  const float* pg_down_w = (const float*)d_in[14];
  const float* pg_up_w   = (const float*)d_in[15];
  const float* pg_prompt_w = (const float*)d_in[16];
  const float* prompt_param = (const float*)d_in[17];
  const float* pg_q_w    = (const float*)d_in[18];
  const float* pg_kv_w   = (const float*)d_in[19];
  const float* pg_proj_w = (const float*)d_in[20];
  const float* pg_proj_b = (const float*)d_in[21];
  const float* fc1_w     = (const float*)d_in[22];
  const float* fc1_b     = (const float*)d_in[23];
  const float* fc2_w     = (const float*)d_in[24];
  const float* fc2_b     = (const float*)d_in[25];
  float* out = (float*)d_out;

  char* ws = (char*)d_ws;
  size_t off = 0;
  auto alloc = [&](size_t bytes) -> char* {
    char* p = ws + off;
    off += (bytes + 255) & ~(size_t)255;
    return p;
  };

  bf16* w_qkv    = (bf16*)alloc((size_t)T3C_ * C_ * 2);
  bf16* w_proj   = (bf16*)alloc((size_t)C_ * C_ * 2);
  bf16* w_gsqkv  = (bf16*)alloc((size_t)T3C_ * C_ * 2);
  bf16* w_gsproj = (bf16*)alloc((size_t)C_ * C_ * 2);
  bf16* w_fc1    = (bf16*)alloc((size_t)FC1O_ * C_ * 2);
  bf16* w_fc2    = (bf16*)alloc((size_t)C_ * HID_ * 2);
  float* norms   = (float*)alloc((size_t)B_ * 384 * 4);
  bf16* a2bf     = (bf16*)alloc((size_t)B_ * HEADS_ * 1024 * 2);
  float* xm      = (float*)alloc((size_t)B_ * C_ * 4);
  float* pgvec   = (float*)alloc((size_t)B_ * C_ * 4);
  bf16* tokbf    = (bf16*)alloc((size_t)MTOT_ * C_ * 2);
  bf16* rA       = (bf16*)alloc((size_t)MTOT_ * C_ * 2);    // xw / attno / ot / xln2
  bf16* rBig     = (bf16*)alloc((size_t)MTOT_ * HID_ * 2);  // qkv(576) / qkv2t(576) / gated(768)
  float* xsp     = (float*)alloc((size_t)MTOT_ * C_ * 4);
  float* specx   = (float*)alloc((size_t)MTOT_ * C_ * 4);   // spec, then x in-place
  bf16* qkchw    = (bf16*)alloc((size_t)B_ * 384 * N_ * 2);
  bf16* v2t      = (bf16*)alloc((size_t)MTOT_ * C_ * 2);

  // weight conversions
  auto cvt = [&](const float* src, bf16* dst, int n) {
    k_cvt<<<(n + 255) / 256, 256, 0, stream>>>(src, dst, n);
  };
  cvt(qkv_w,    w_qkv,    T3C_ * C_);
  cvt(proj_w,   w_proj,   C_ * C_);
  cvt(gs_qkv_w, w_gsqkv,  T3C_ * C_);
  cvt(gs_proj_w,w_gsproj, C_ * C_);
  cvt(fc1_w,    w_fc1,    FC1O_ * C_);
  cvt(fc2_w,    w_fc2,    C_ * HID_);

  // 1) LN1 + windowing + token-major bf16 copy
  k_prep<<<MTOT_ / 8, 256, 0, stream>>>(in_put, ln1_g, ln1_b, tokbf, rA);
  // 2) QKV GEMM (windowed rows)
  k_gemm<OUT_BF16, true><<<dim3(MTOT_ / 128, T3C_ / 64), 256, 0, stream>>>(
      rA, w_qkv, qkv_b, rBig, MTOT_, T3C_, C_);
  // 3) windowed attention
  k_swin<<<1024 * HEADS_, 128, 0, stream>>>(rBig, rpb_table, rA);
  // 4) attention proj -> xsp (token order)
  k_gemm<OUT_XSP, true><<<dim3(MTOT_ / 128, C_ / 64), 256, 0, stream>>>(
      rA, w_proj, proj_b, xsp, MTOT_, C_, C_);
  // 5) spectral 1x1 conv (qkv2, token-major)
  k_gemm<OUT_BF16, false><<<dim3(MTOT_ / 128, T3C_ / 64), 256, 0, stream>>>(
      tokbf, w_gsqkv, nullptr, rBig, MTOT_, T3C_, C_);
  // 6) depthwise 3x3
  k_dwconv<<<MTOT_, 576, 0, stream>>>(rBig, gs_dw_w, qkchw, v2t);
  // 7) row norms of q2/k2
  k_rownorm<<<B_ * 384, 256, 0, stream>>>(qkchw, norms);
  // 8) channel attention a2 (per b,head)
  k_a2<<<B_ * HEADS_, 256, 0, stream>>>(qkchw, norms, temperature, a2bf);
  // 9) o2^T token-major
  k_o2t<<<B_ * HEADS_ * (N_ / 128), 128, 0, stream>>>(v2t, a2bf, rA);
  // 10) spectral proj -> spec (f32, token order)
  k_gemm<OUT_F32, false><<<dim3(MTOT_ / 128, C_ / 64), 256, 0, stream>>>(
      rA, w_gsproj, nullptr, specx, MTOT_, C_, C_);
  // 11) mean over N
  k_mean<<<B_ * C_, 256, 0, stream>>>(specx, xm);
  // 12) prompt-guided gate vector
  k_pg<<<1, 192, 0, stream>>>(xm, pg_prompt_w, pg_down_w, prompt_param, pg_q_w,
                              pg_kv_w, pg_proj_w, pg_proj_b, pg_up_w, pgvec);
  // 13) residual merge + LN2 (x overwrites spec buffer in-place)
  k_merge<<<MTOT_ / 8, 256, 0, stream>>>(in_put, xsp, specx, pgvec, ln2_g, ln2_b, rA);
  // 14) fc1 + fused GeLU gate
  k_fc1gate<<<dim3(MTOT_ / 128, HID_ / 64), 256, 0, stream>>>(rA, w_fc1, fc1_b, rBig);
  // 15) fc2 + bias + residual + transpose to (B,C,H,W)
  k_fc2<<<dim3(MTOT_ / 128, C_ / 64), 256, 0, stream>>>(rBig, w_fc2, fc2_b, specx, out);
}
